// CausalSelfAttention_22917945491489
// MI455X (gfx1250) — matrix operand order repaired and checked by reading, not compiled, not run on hardware
//
#include <hip/hip_runtime.h>

// ---------------------------------------------------------------------------
// CDNA5 (gfx1250) causal self-attention forward, bf16 WMMA path.
//   B=4, T=2048, C=2048, H=16, hs=128
// ---------------------------------------------------------------------------

typedef __attribute__((ext_vector_type(16))) __bf16 v16bf;
typedef __attribute__((ext_vector_type(8)))  __bf16 v8bf;
typedef __attribute__((ext_vector_type(8)))  float  v8f;

#define B_   4
#define T_   2048
#define C_   2048
#define H_   16
#define D_   128
#define BT_  (B_ * T_)          // 8192
#define NELEM ((size_t)B_ * H_ * T_ * D_)   // 16,777,216

// ---------------- bf16 helpers (bit-exact RNE convert) ----------------------
__device__ __forceinline__ __bf16 f2bf(float f) {
  unsigned u = __builtin_bit_cast(unsigned, f);
  unsigned r = u + 0x7FFFu + ((u >> 16) & 1u);
  unsigned short h = (unsigned short)(r >> 16);
  return __builtin_bit_cast(__bf16, h);
}
__device__ __forceinline__ float bf2f(__bf16 b) {
  unsigned short h = __builtin_bit_cast(unsigned short, b);
  unsigned u = ((unsigned)h) << 16;
  return __builtin_bit_cast(float, u);
}
__device__ __forceinline__ v8f vzero8() {
  v8f v;
#pragma unroll
  for (int i = 0; i < 8; ++i) v[i] = 0.f;
  return v;
}

// ---------------- WMMA fragment loaders -------------------------------------
// A fragment (16x32 bf16, row-major source): lane row = l%16, hf = l/16.
// element e (0..15): K = hf*16 + e
__device__ __forceinline__ v16bf load_a16(const __bf16* base, int hf) {
  v8bf lo = *reinterpret_cast<const v8bf*>(base + hf * 16);
  v8bf hi = *reinterpret_cast<const v8bf*>(base + hf * 16 + 8);
  v16bf a;
#pragma unroll
  for (int i = 0; i < 8; ++i) { a[i] = lo[i]; a[8 + i] = hi[i]; }
  return a;
}
// B fragment (32x16 bf16): lane col = l%16, element e: K = (l/16)*16 + e.
// Caller passes pointer already offset by (l/16)*16 along K; 16 contiguous.
__device__ __forceinline__ v16bf load_b16(const __bf16* base) {
  v8bf lo = *reinterpret_cast<const v8bf*>(base);
  v8bf hi = *reinterpret_cast<const v8bf*>(base + 8);
  v16bf b;
#pragma unroll
  for (int i = 0; i < 8; ++i) { b[i] = lo[i]; b[8 + i] = hi[i]; }
  return b;
}
__device__ __forceinline__ v8f wmma_bf16(v16bf a, v16bf b, v8f c) {
  return __builtin_amdgcn_wmma_f32_16x16x32_bf16(false, a, false, b,
                                                 (short)0, c, false, false);
}

// ---------------- cross-lane reductions over 16-lane halves -----------------
__device__ __forceinline__ float redmax16(float v) {
#pragma unroll
  for (int m = 1; m < 16; m <<= 1) v = fmaxf(v, __shfl_xor(v, m, 32));
  return v;
}
__device__ __forceinline__ float redsum16(float v) {
#pragma unroll
  for (int m = 1; m < 16; m <<= 1) v += __shfl_xor(v, m, 32);
  return v;
}

// ---------------- pack / convert kernels ------------------------------------
__global__ void k_convert_bf16(const float* __restrict__ in,
                               __bf16* __restrict__ out, size_t n) {
  for (size_t i = (size_t)blockIdx.x * blockDim.x + threadIdx.x; i < n;
       i += (size_t)gridDim.x * blockDim.x)
    out[i] = f2bf(in[i]);
}

// Wt[n][k] = bf16(W[k][n])  — LDS-tiled 32x32 transpose, coalesced both sides
__global__ __launch_bounds__(256) void k_pack_wt(const float* __restrict__ W,
                                                 __bf16* __restrict__ Wt,
                                                 int K, int N) {
  __shared__ __bf16 tile[32][33];
  const int k0 = blockIdx.x * 32, n0 = blockIdx.y * 32;
  const int tx = threadIdx.x & 31, ty = threadIdx.x >> 5;
#pragma unroll
  for (int r = 0; r < 32; r += 8)
    tile[ty + r][tx] = f2bf(W[(size_t)(k0 + ty + r) * N + n0 + tx]);
  __syncthreads();
#pragma unroll
  for (int r = 0; r < 32; r += 8)
    Wt[(size_t)(n0 + ty + r) * K + k0 + tx] = tile[tx][ty + r];
}

__global__ void k_init_scales(unsigned* s) {
  if (threadIdx.x < 8) s[threadIdx.x] = 0u;
}

__global__ void k_absmax(const __bf16* __restrict__ p, size_t n,
                         unsigned* __restrict__ slot) {
  float m = 0.f;
  for (size_t i = (size_t)blockIdx.x * blockDim.x + threadIdx.x; i < n;
       i += (size_t)gridDim.x * blockDim.x)
    m = fmaxf(m, fabsf(bf2f(p[i])));
#pragma unroll
  for (int s = 1; s < 32; s <<= 1) m = fmaxf(m, __shfl_xor(m, s, 32));
  if ((threadIdx.x & 31) == 0) atomicMax(slot, __float_as_uint(m));
}

// per-tensor int8 quant-dequant (K path, layout preserved)
__global__ void k_qd(const __bf16* __restrict__ in, __bf16* __restrict__ out,
                     const unsigned* __restrict__ slot) {
  float am = __uint_as_float(*slot);
  float scale = am > 0.f ? am * (1.f / 127.f) : 1.f;
  for (size_t i = (size_t)blockIdx.x * blockDim.x + threadIdx.x; i < NELEM;
       i += (size_t)gridDim.x * blockDim.x) {
    float v = bf2f(in[i]);
    float q = fminf(fmaxf(rintf(v / scale), -127.f), 127.f);
    out[i] = f2bf(q * scale);
  }
}

// V path: quant-dequant + LDS-tiled transpose [bh][T][D] -> [bh][D][T]
__global__ __launch_bounds__(256) void k_qd_vt(const __bf16* __restrict__ in,
                                               __bf16* __restrict__ outT,
                                               const unsigned* __restrict__ slot) {
  __shared__ __bf16 tile[32][33];
  float am = __uint_as_float(*slot);
  float scale = am > 0.f ? am * (1.f / 127.f) : 1.f;
  const int t0 = blockIdx.x * 32, d0 = blockIdx.y * 32, bh = blockIdx.z;
  const int tx = threadIdx.x & 31, ty = threadIdx.x >> 5;
#pragma unroll
  for (int r = 0; r < 32; r += 8) {
    float v = bf2f(in[((size_t)bh * T_ + t0 + ty + r) * D_ + d0 + tx]);
    float q = fminf(fmaxf(rintf(v / scale), -127.f), 127.f);
    tile[ty + r][tx] = f2bf(q * scale);
  }
  __syncthreads();
#pragma unroll
  for (int r = 0; r < 32; r += 8)
    outT[((size_t)bh * D_ + d0 + ty + r) * T_ + t0 + tx] = tile[tx][ty + r];
}

// ---------------- GEMM 1: qkv = Xbf @ W_attn, scatter to Q/K/V --------------
// Wave tile 32(M) x 64(N): 2 A-frags, 4 B-frags, 8 WMMAs per k32 step.
__global__ __launch_bounds__(256) void k_gemm_qkv(
    const __bf16* __restrict__ X, const __bf16* __restrict__ Wt,
    __bf16* __restrict__ Qb, __bf16* __restrict__ Kb, __bf16* __restrict__ Vb) {
  const int lane = threadIdx.x & 31, wave = threadIdx.x >> 5;
  const int hf = lane >> 4, ln = lane & 15;
  const int m0 = blockIdx.x * 256 + wave * 32;
  const int n0 = blockIdx.y * 64;
  const __bf16* arow0 = X + (size_t)(m0 + ln) * C_;
  const __bf16* arow1 = X + (size_t)(m0 + 16 + ln) * C_;
  v8f acc[2][4];
#pragma unroll
  for (int mt = 0; mt < 2; ++mt)
#pragma unroll
    for (int nt = 0; nt < 4; ++nt) acc[mt][nt] = vzero8();
  for (int k0 = 0; k0 < C_; k0 += 32) {
    v16bf a0 = load_a16(arow0 + k0, hf);
    v16bf a1 = load_a16(arow1 + k0, hf);
#pragma unroll
    for (int nt = 0; nt < 4; ++nt) {
      const __bf16* bp = Wt + (size_t)(n0 + nt * 16 + ln) * C_ + k0 + hf * 16;
      v16bf b = load_b16(bp);
      acc[0][nt] = wmma_bf16(a0, b, acc[0][nt]);
      acc[1][nt] = wmma_bf16(a1, b, acc[1][nt]);
    }
  }
#pragma unroll
  for (int mt = 0; mt < 2; ++mt)
#pragma unroll
    for (int nt = 0; nt < 4; ++nt) {
      int n = n0 + nt * 16 + ln;
      int which = n >> 11, nc = n & (C_ - 1);
      int h = nc >> 7, d = nc & (D_ - 1);
      __bf16* dst = which == 0 ? Qb : (which == 1 ? Kb : Vb);
#pragma unroll
      for (int r = 0; r < 8; ++r) {
        int m = m0 + mt * 16 + r + 8 * hf;
        int bb = m >> 11, t = m & (T_ - 1);
        dst[(((size_t)bb * H_ + h) * T_ + t) * D_ + d] = f2bf(acc[mt][nt][r]);
      }
    }
}

// ---------------- Flash-style causal SDPA (one wave = 16 query rows) --------
__global__ __launch_bounds__(128) void k_attn(
    const __bf16* __restrict__ Qb, const __bf16* __restrict__ Kq,
    const __bf16* __restrict__ Vt, __bf16* __restrict__ Yb) {
  __shared__ __align__(16) __bf16 Pbuf[4][16][32];
  const int lane = threadIdx.x & 31, wave = threadIdx.x >> 5;
  const int hf = lane >> 4, ln = lane & 15;
  const int h = blockIdx.y, b = blockIdx.z;
  const int bh = b * H_ + h;
  const int t0 = (blockIdx.x * 4 + wave) * 16;
  const float sc = 0.08838834764831845f;  // 1/sqrt(128)

  // preload Q fragments for the whole 16x128 block
  const __bf16* qbase = Qb + ((size_t)bh * T_ + t0 + ln) * D_;
  v16bf aq[4];
#pragma unroll
  for (int kb = 0; kb < 4; ++kb) aq[kb] = load_a16(qbase + kb * 32, hf);

  v8f o[8];
#pragma unroll
  for (int dd = 0; dd < 8; ++dd) o[dd] = vzero8();
  float m_run[8], l_run[8];
#pragma unroll
  for (int r = 0; r < 8; ++r) { m_run[r] = -INFINITY; l_run[r] = 0.f; }

  for (int j0 = 0; j0 < t0 + 16; j0 += 32) {
    // ---- S = Q @ K^T  (two 16x16 key subtiles, K-dim = hs = 128) ----
    v8f s0 = vzero8(), s1 = vzero8();
#pragma unroll
    for (int kb = 0; kb < 4; ++kb) {
      const __bf16* k0p =
          Kq + ((size_t)bh * T_ + j0 + ln) * D_ + kb * 32 + hf * 16;
      const __bf16* k1p =
          Kq + ((size_t)bh * T_ + j0 + 16 + ln) * D_ + kb * 32 + hf * 16;
      s0 = wmma_bf16(aq[kb], load_b16(k0p), s0);
      s1 = wmma_bf16(aq[kb], load_b16(k1p), s1);
    }
    // ---- online softmax over C-layout rows (row = r + 8*hf, col = ln) ----
    float f[8], p0[8], p1[8];
#pragma unroll
    for (int r = 0; r < 8; ++r) {
      int qi = t0 + r + 8 * hf;
      float x0 = (j0 + ln <= qi) ? s0[r] * sc : -INFINITY;
      float x1 = (j0 + 16 + ln <= qi) ? s1[r] * sc : -INFINITY;
      float mt = redmax16(fmaxf(x0, x1));
      float mn = fmaxf(m_run[r], mt);
      float e0 = __expf(x0 - mn);
      float e1 = __expf(x1 - mn);
      float fr = __expf(m_run[r] - mn);
      l_run[r] = l_run[r] * fr + redsum16(e0 + e1);
      m_run[r] = mn;
      f[r] = fr; p0[r] = e0; p1[r] = e1;
    }
#pragma unroll
    for (int dd = 0; dd < 8; ++dd)
#pragma unroll
      for (int r = 0; r < 8; ++r) o[dd][r] *= f[r];

    // ---- bounce P (C-layout -> A-fragment layout) through LDS ----
#pragma unroll
    for (int r = 0; r < 8; ++r) {
      Pbuf[wave][r + 8 * hf][ln] = f2bf(p0[r]);
      Pbuf[wave][r + 8 * hf][16 + ln] = f2bf(p1[r]);
    }
    asm volatile("s_wait_dscnt 0" ::: "memory");  // DS in-order within wave
    v16bf ap = load_a16(&Pbuf[wave][ln][0], hf);

    // ---- O += P @ V  (V pre-transposed: Vt[bh][d][t]) ----
#pragma unroll
    for (int dd = 0; dd < 8; ++dd) {
      const __bf16* vp =
          Vt + ((size_t)bh * D_ + dd * 16 + ln) * T_ + j0 + hf * 16;
      o[dd] = wmma_bf16(ap, load_b16(vp), o[dd]);
    }
  }

  // ---- normalize and write back to (B,T,H*D) row-major bf16 ----
  float inv[8];
#pragma unroll
  for (int r = 0; r < 8; ++r) inv[r] = 1.f / l_run[r];
#pragma unroll
  for (int dd = 0; dd < 8; ++dd)
#pragma unroll
    for (int r = 0; r < 8; ++r) {
      size_t row = (size_t)b * T_ + t0 + r + 8 * hf;
      Yb[row * C_ + h * D_ + dd * 16 + ln] = f2bf(o[dd][r] * inv[r]);
    }
}

// ---------------- GEMM 2: out = Yb @ W_proj (fp32 out) ----------------------
__global__ __launch_bounds__(256) void k_gemm_proj(
    const __bf16* __restrict__ Y, const __bf16* __restrict__ Wt,
    float* __restrict__ out) {
  const int lane = threadIdx.x & 31, wave = threadIdx.x >> 5;
  const int hf = lane >> 4, ln = lane & 15;
  const int m0 = blockIdx.x * 256 + wave * 32;
  const int n0 = blockIdx.y * 64;
  const __bf16* arow0 = Y + (size_t)(m0 + ln) * C_;
  const __bf16* arow1 = Y + (size_t)(m0 + 16 + ln) * C_;
  v8f acc[2][4];
#pragma unroll
  for (int mt = 0; mt < 2; ++mt)
#pragma unroll
    for (int nt = 0; nt < 4; ++nt) acc[mt][nt] = vzero8();
  for (int k0 = 0; k0 < C_; k0 += 32) {
    v16bf a0 = load_a16(arow0 + k0, hf);
    v16bf a1 = load_a16(arow1 + k0, hf);
#pragma unroll
    for (int nt = 0; nt < 4; ++nt) {
      const __bf16* bp = Wt + (size_t)(n0 + nt * 16 + ln) * C_ + k0 + hf * 16;
      v16bf b = load_b16(bp);
      acc[0][nt] = wmma_bf16(a0, b, acc[0][nt]);
      acc[1][nt] = wmma_bf16(a1, b, acc[1][nt]);
    }
  }
#pragma unroll
  for (int mt = 0; mt < 2; ++mt)
#pragma unroll
    for (int nt = 0; nt < 4; ++nt)
#pragma unroll
      for (int r = 0; r < 8; ++r)
        out[(size_t)(m0 + mt * 16 + r + 8 * hf) * C_ + n0 + nt * 16 + ln] =
            acc[mt][nt][r];
}

// ---------------------------------------------------------------------------
extern "C" void kernel_launch(void* const* d_in, const int* in_sizes, int n_in,
                              void* d_out, int out_size, void* d_ws,
                              size_t ws_size, hipStream_t stream) {
  const float* x  = (const float*)d_in[0];   // (4,2048,2048)
  const float* Wa = (const float*)d_in[1];   // (2048,6144)
  const float* Wp = (const float*)d_in[2];   // (2048,2048)
  float* out = (float*)d_out;                // (4,2048,2048)

  char* w = (char*)d_ws;
  auto carve = [&](size_t bytes) -> void* {
    void* p = (void*)w;
    w += (bytes + 255) & ~(size_t)255;
    return p;
  };
  const size_t half_mb = (size_t)BT_ * C_ * 2;          // 32 MiB bf16 planes
  __bf16* Xbf = (__bf16*)carve(half_mb);                // [8192][2048]
  __bf16* Wat = (__bf16*)carve((size_t)3 * C_ * C_ * 2);// [6144][2048] (N x K)
  __bf16* Wpt = (__bf16*)carve((size_t)C_ * C_ * 2);    // [2048][2048] (N x K)
  __bf16* Qb  = (__bf16*)carve(half_mb);                // [B][H][T][D]
  __bf16* Kb  = (__bf16*)carve(half_mb);                // [B][H][T][D]
  __bf16* Vb  = (__bf16*)carve(half_mb);                // [B][H][T][D]
  __bf16* Kq  = (__bf16*)carve(half_mb);                // quantized K
  __bf16* Vt  = (__bf16*)carve(half_mb);                // [B][H][D][T]
  __bf16* Yb  = (__bf16*)carve(half_mb);                // attn out (B,T,C)
  unsigned* scales = (unsigned*)carve(256);

  k_init_scales<<<1, 32, 0, stream>>>(scales);
  k_convert_bf16<<<4096, 256, 0, stream>>>(x, Xbf, (size_t)BT_ * C_);
  k_pack_wt<<<dim3(C_ / 32, (3 * C_) / 32), 256, 0, stream>>>(Wa, Wat, C_,
                                                              3 * C_);
  k_pack_wt<<<dim3(C_ / 32, C_ / 32), 256, 0, stream>>>(Wp, Wpt, C_, C_);

  k_gemm_qkv<<<dim3(BT_ / 256, (3 * C_) / 64), 256, 0, stream>>>(Xbf, Wat, Qb,
                                                                 Kb, Vb);

  k_absmax<<<1024, 256, 0, stream>>>(Kb, NELEM, scales + 0);
  k_absmax<<<1024, 256, 0, stream>>>(Vb, NELEM, scales + 1);
  k_qd<<<4096, 256, 0, stream>>>(Kb, Kq, scales + 0);
  k_qd_vt<<<dim3(T_ / 32, D_ / 32, B_ * H_), 256, 0, stream>>>(Vb, Vt,
                                                               scales + 1);

  k_attn<<<dim3(T_ / 64, H_, B_), 128, 0, stream>>>(Qb, Kq, Vt, Yb);

  k_gemm_proj<<<dim3(BT_ / 256, C_ / 64), 256, 0, stream>>>(Yb, Wpt, out);
}
